// SparseBlockLoader_74088185856324
// MI455X (gfx1250) — matrix operand order repaired and checked by reading, not compiled, not run on hardware
//
#include <hip/hip_runtime.h>

// ---------------------------------------------------------------------------
// MoE block (pooled routing, top-2 of 16 experts) + LM head for gfx1250.
// bf16 WMMA (v_wmma_f32_16x16x32_bf16) with f32 accumulation.
// A tiles moved by the Tensor Data Mover (tensor_load_to_lds, TENSORcnt),
// double-buffered LDS so TDM/B-staging overlap the WMMA stream.
// ---------------------------------------------------------------------------

typedef __bf16 bf16_t;
typedef __attribute__((ext_vector_type(16))) __bf16 v16bf;
typedef __attribute__((ext_vector_type(8)))  __bf16 v8bf;
typedef __attribute__((ext_vector_type(8)))  float  v8f;

#define T_TOK 8192   // B*L = 4*2048
#define DIM   1024
#define HID   4096
#define VOC   32000
#define NEXP  16

#define TM 128
#define TN 128
#define TK 32
#define LDS_STRIDE (TK + 8)   // 40 bf16 -> 80B rows, 16B aligned

enum { MODE_GELU_BF16 = 0, MODE_SCALE_ACC_F32 = 1, MODE_STORE_F32 = 2 };

#if defined(__gfx1250__) && __has_builtin(__builtin_amdgcn_tensor_load_to_lds)
#define HAS_TDM 1
#else
#define HAS_TDM 0
#endif

#if HAS_TDM
typedef __attribute__((ext_vector_type(4))) unsigned int u32x4;
typedef __attribute__((ext_vector_type(4))) int          i32x4;
typedef __attribute__((ext_vector_type(8))) int          i32x8;

// TDM: DMA a TM x TK bf16 tile (row stride K elements) from global into LDS,
// padding each 64B tile row out to the 80B LDS stride. D# per ISA 8.3/8.4.
__device__ __forceinline__ void tdm_load_a_tile(const bf16_t* gptr,
                                                unsigned lds_addr, int K) {
  const unsigned long long ga = (unsigned long long)(uintptr_t)gptr;
  u32x4 g0;
  g0[0] = 1u;                                           // count=1, user D#
  g0[1] = lds_addr;                                     // lds_addr
  g0[2] = (unsigned)ga;                                 // global_addr[31:0]
  g0[3] = (unsigned)((ga >> 32) & 0x01ffffffu) | (2u << 30); // [56:32] | type=2
  i32x8 g1;
  g1[0] = (1 << 16)      // data_size = 2 bytes
        | (1 << 20)      // pad_enable
        | (3 << 22)      // pad_interval: 16 DWORDs (64B) between pads
        | (3 << 25);     // pad_amount:   4 DWORDs (16B) of pad
  g1[1] = (K & 0xffff) << 16;                           // tensor_dim0[15:0]
  g1[2] = ((unsigned)K >> 16) | ((T_TOK & 0xffff) << 16); // dim0 hi | dim1 lo
  g1[3] = (TK << 16);                                   // tile_dim0 = 32
  g1[4] = TM;                                           // tile_dim1=128, dim2=0
  g1[5] = K;                                            // tensor_dim0_stride lo
  g1[6] = 0;
  g1[7] = 0;
  const i32x4 z4 = {0, 0, 0, 0};
#if __clang_major__ >= 23
  const i32x8 z8 = {0, 0, 0, 0, 0, 0, 0, 0};
  __builtin_amdgcn_tensor_load_to_lds(g0, g1, z4, z4, z8, 0);
#else
  __builtin_amdgcn_tensor_load_to_lds(g0, g1, z4, z4, 0);
#endif
}
#endif  // HAS_TDM

// ---------------------------- small kernels --------------------------------

__global__ __launch_bounds__(256) void zero_pool_kernel(float* __restrict__ pool) {
  const int i = threadIdx.x;
#pragma unroll
  for (int r = 0; r < 4; ++r) pool[i + r * 256] = 0.0f;
}

// pool[d] += sum over 512 tokens of embed[id[t]][d]; grid (DIM/256, T_TOK/512)
__global__ __launch_bounds__(256) void pool_accum_kernel(const int* __restrict__ ids,
                                                         const float* __restrict__ emb,
                                                         float* __restrict__ pool) {
  const int d  = blockIdx.x * 256 + threadIdx.x;
  const int t0 = blockIdx.y * 512;
  float s = 0.0f;
  for (int i = 0; i < 512; ++i) {
    const int id = ids[t0 + i];
    s += emb[(size_t)id * DIM + d];
  }
  atomicAdd(&pool[d], s);
}

// x_bf16[token][:] = bf16(embed[id[token]][:]); one block per token
__global__ __launch_bounds__(256) void gather_embed_kernel(const int* __restrict__ ids,
                                                           const float* __restrict__ emb,
                                                           bf16_t* __restrict__ xb) {
  const int token = blockIdx.x;
  const int id = ids[token];
  const float4 v = ((const float4*)(emb + (size_t)id * DIM))[threadIdx.x];
  union { bf16_t h[4]; uint2 u; } p;
  p.h[0] = (bf16_t)v.x; p.h[1] = (bf16_t)v.y;
  p.h[2] = (bf16_t)v.z; p.h[3] = (bf16_t)v.w;
  ((uint2*)(xb + (size_t)token * DIM))[threadIdx.x] = p.u;
}

// Router: logits = (pool_sum/T) @ w_router, softmax, top-2, renormalize.
__global__ void router_kernel(const float* __restrict__ pool,
                              const float* __restrict__ w_router,
                              float* __restrict__ gate,
                              int* __restrict__ gidx) {
  __shared__ float logits[NEXP];
  const int t = threadIdx.x;
  if (t < NEXP) {
    float s = 0.0f;
    for (int d = 0; d < DIM; ++d) s += pool[d] * w_router[d * NEXP + t];
    logits[t] = s * (1.0f / (float)T_TOK);
  }
  __syncthreads();
  if (t == 0) {
    int i0 = 0;
    for (int n = 1; n < NEXP; ++n) if (logits[n] > logits[i0]) i0 = n;
    int i1 = (i0 == 0) ? 1 : 0;
    for (int n = 0; n < NEXP; ++n)
      if (n != i0 && logits[n] > logits[i1]) i1 = n;
    const float l0 = logits[i0], l1 = logits[i1];
    const float e1 = __expf(l1 - l0);       // renormalized top-2 softmax
    const float w0 = 1.0f / (1.0f + e1);
    gate[0] = w0; gate[1] = 1.0f - w0;
    gidx[0] = i0; gidx[1] = i1;
  }
}

__global__ __launch_bounds__(256) void f32_to_bf16_kernel(const float* __restrict__ in,
                                                          bf16_t* __restrict__ out) {
  const size_t i = (size_t)blockIdx.x * 256 + threadIdx.x;
  const float4 v = ((const float4*)in)[i];
  union { bf16_t h[4]; uint2 u; } p;
  p.h[0] = (bf16_t)v.x; p.h[1] = (bf16_t)v.y;
  p.h[2] = (bf16_t)v.z; p.h[3] = (bf16_t)v.w;
  ((uint2*)out)[i] = p.u;
}

// ---------------------------- WMMA GEMM ------------------------------------

__device__ __forceinline__ float gelu_tanh(float x) {
  const float c0 = 0.7978845608028654f;   // sqrt(2/pi)
  const float c1 = 0.044715f;
  return 0.5f * x * (1.0f + tanhf(c0 * (x + c1 * x * x * x)));
}

union U16frag { v16bf v; v8bf h[2]; };

// C[M x N] = A[M x K](bf16) @ B[K x N](f32, converted to bf16 on load)
// 256 threads = 8 wave32; block tile 128x128; each wave 64x32 (4x2 WMMA accs).
// Double-buffered LDS: TDM streams the next A tile while WMMAs consume this one.
__global__ __launch_bounds__(256, 2)
void gemm_bf16_wmma_kernel(const bf16_t* __restrict__ A,
                           const float* __restrict__ Bbase,
                           size_t expert_stride,
                           const float* __restrict__ gate,
                           const int* __restrict__ gidx,
                           int slot,                 // expert slot 0/1, or -1
                           bf16_t* __restrict__ Cbf, // MODE_GELU_BF16
                           float* __restrict__ Cf,   // f32 modes
                           int N, int K,
                           int mode, int accumulate) {
  __shared__ bf16_t As[2][TM][LDS_STRIDE];   // [buf][row][k]
  __shared__ bf16_t Bs[2][TN][LDS_STRIDE];   // [buf][col][k] (transposed)

  const int tid  = threadIdx.x;
  const int lane = tid & 31;
  const int wave = tid >> 5;
  const int wm   = wave >> 2;      // 0..1 -> 64 rows
  const int wn   = wave & 3;       // 0..3 -> 32 cols
  const int lh   = lane & 15;
  const int hsel = lane >> 4;      // half-wave select per WMMA layout

  const int m0 = blockIdx.x * TM;
  const int n0 = blockIdx.y * TN;

  const float* __restrict__ B =
      Bbase + ((slot >= 0) ? (size_t)gidx[slot] * expert_stride : (size_t)0);

  // B staging map: each thread owns a 4(k) x 4(col) micro-block.
  const int bkg = (tid >> 5) * 4;   // k base within tile (0..28)
  const int bcg = (tid & 31) * 4;   // col base within tile (0..124)
  const float* bptr = B + (size_t)bkg * N + (n0 + bcg);

#if !HAS_TDM
  // fallback A staging map: 16 bf16 per thread via two b128 loads
  const int arow = tid >> 1;
  const int acol = (tid & 1) * 16;
  const bf16_t* aptr = A + (size_t)(m0 + arow) * K + acol;
#endif

  // stage B tile (f32 -> bf16, register 4x4 transpose, packed b64 LDS stores)
  auto stage_b = [&](int b, int kk) {
    const float* p = bptr + (size_t)kk * N;
    const float4 r0 = *(const float4*)(p);
    const float4 r1 = *(const float4*)(p + N);
    const float4 r2 = *(const float4*)(p + 2 * N);
    const float4 r3 = *(const float4*)(p + 3 * N);
    const float rr[4][4] = {{r0.x, r0.y, r0.z, r0.w},
                            {r1.x, r1.y, r1.z, r1.w},
                            {r2.x, r2.y, r2.z, r2.w},
                            {r3.x, r3.y, r3.z, r3.w}};
#pragma unroll
    for (int j = 0; j < 4; ++j) {
      union { bf16_t h[4]; uint2 u; } pk;
#pragma unroll
      for (int i = 0; i < 4; ++i) pk.h[i] = (bf16_t)rr[i][j];
      *(uint2*)&Bs[b][bcg + j][bkg] = pk.u;
    }
  };

  auto stage_a = [&](int b, int kk) {
#if HAS_TDM
    if (tid < 32)   // one wave issues the DMA for the whole tile
      tdm_load_a_tile(A + (size_t)m0 * K + kk,
                      (unsigned)(uintptr_t)&As[b][0][0], K);
#else
    const bf16_t* p = aptr + kk;
    const uint4 a0 = *(const uint4*)(p);
    const uint4 a1 = *(const uint4*)(p + 8);
    *(uint4*)&As[b][arow][acol]     = a0;
    *(uint4*)&As[b][arow][acol + 8] = a1;
#endif
  };

  const v8f vzero = {0.f, 0.f, 0.f, 0.f, 0.f, 0.f, 0.f, 0.f};
  v8f acc[4][2];
#pragma unroll
  for (int mt = 0; mt < 4; ++mt)
#pragma unroll
    for (int nt = 0; nt < 2; ++nt) acc[mt][nt] = vzero;

  // ---- prologue: fill buffer 0 ----
  stage_a(0, 0);
  stage_b(0, 0);
#if HAS_TDM
  __builtin_amdgcn_s_wait_tensorcnt(0);
#endif
  __syncthreads();

  for (int k0 = 0; k0 < K; k0 += TK) {
    const int buf = (k0 / TK) & 1;
    const int nxt = buf ^ 1;

    // ---- stage next tiles into the other buffer (overlaps the WMMAs) ----
    if (k0 + TK < K) {
      stage_a(nxt, k0 + TK);
      stage_b(nxt, k0 + TK);
      if (k0 + 2 * TK < K)
        __builtin_prefetch(bptr + (size_t)(k0 + 2 * TK) * N, 0, 3);
    }

    // ---- build fragments (ds_load_b128) ----
    v16bf afrag[4];
#pragma unroll
    for (int mt = 0; mt < 4; ++mt) {
      const int row = wm * 64 + mt * 16 + lh;
      const int kh  = hsel * 8;
      U16frag u;
      u.h[0] = *(const v8bf*)&As[buf][row][kh];        // K = kh..kh+7
      u.h[1] = *(const v8bf*)&As[buf][row][kh + 16];   // K = 16+kh..
      afrag[mt] = u.v;
    }
    v16bf bfrag[2];
#pragma unroll
    for (int nt = 0; nt < 2; ++nt) {
      const int col = wn * 32 + nt * 16 + lh;
      const int kb  = hsel * 8;
      U16frag u;
      u.h[0] = *(const v8bf*)&Bs[buf][col][kb];
      u.h[1] = *(const v8bf*)&Bs[buf][col][kb + 16];
      bfrag[nt] = u.v;
    }

    // ---- 8x v_wmma_f32_16x16x32_bf16 ----
#pragma unroll
    for (int mt = 0; mt < 4; ++mt)
#pragma unroll
      for (int nt = 0; nt < 2; ++nt)
        acc[mt][nt] = __builtin_amdgcn_wmma_f32_16x16x32_bf16(
            false, afrag[mt], false, bfrag[nt],
            (short)0, acc[mt][nt], false, false);

#if HAS_TDM
    __builtin_amdgcn_s_wait_tensorcnt(0);   // next A tile landed in LDS
#endif
    __syncthreads();
  }

  // ---- epilogue ----
  float scale = 1.0f;
  if (mode == MODE_SCALE_ACC_F32) scale = gate[slot];
#pragma unroll
  for (int mt = 0; mt < 4; ++mt) {
#pragma unroll
    for (int nt = 0; nt < 2; ++nt) {
      const int col = n0 + wn * 32 + nt * 16 + lh;
#pragma unroll
      for (int r = 0; r < 8; ++r) {
        const int row = m0 + wm * 64 + mt * 16 + r + hsel * 8;
        const float v = acc[mt][nt][r];
        const size_t o = (size_t)row * N + col;
        if (mode == MODE_GELU_BF16) {
          Cbf[o] = (bf16_t)gelu_tanh(v);
        } else if (mode == MODE_SCALE_ACC_F32) {
          const float prev = accumulate ? Cf[o] : 0.0f;
          Cf[o] = prev + scale * v;
        } else {
          Cf[o] = v;
        }
      }
    }
  }
}

// ---------------------------- launch ---------------------------------------

extern "C" void kernel_launch(void* const* d_in, const int* in_sizes, int n_in,
                              void* d_out, int out_size, void* d_ws, size_t ws_size,
                              hipStream_t stream) {
  (void)in_sizes; (void)n_in; (void)out_size; (void)ws_size;
  const int*   input_ids = (const int*)d_in[0];
  const float* embed     = (const float*)d_in[1];
  const float* w_router  = (const float*)d_in[2];
  const float* w1        = (const float*)d_in[3];   // (N, D, H)
  const float* w2        = (const float*)d_in[4];   // (N, H, D)
  const float* w_head    = (const float*)d_in[5];   // (D, V)
  float*       logits    = (float*)d_out;           // (T, V)

  // workspace layout
  const size_t X_BYTES = (size_t)T_TOK * DIM * 2;   // 16 MB  bf16 x
  const size_t H_BYTES = (size_t)T_TOK * HID * 2;   // 64 MB  bf16 h (reused per expert)
  char* ws = (char*)d_ws;
  float*  pool = (float*)(ws + 0);                  // 4 KB
  float*  gate = (float*)(ws + 4096);               // 2 f32
  int*    gidx = (int*)(ws + 4096 + 8);             // 2 i32
  bf16_t* xb   = (bf16_t*)(ws + 8192);
  bf16_t* hb   = (bf16_t*)(ws + 8192 + X_BYTES);
  float*  outf = (float*)(ws + 8192 + X_BYTES + H_BYTES);   // 32 MB f32 combine
  bf16_t* outb = xb;   // alias: x is dead once both expert GEMM1s are done

  const dim3 blk(256);

  // 1) routing inputs
  zero_pool_kernel<<<1, 256, 0, stream>>>(pool);
  pool_accum_kernel<<<dim3(DIM / 256, T_TOK / 512), blk, 0, stream>>>(input_ids, embed, pool);
  gather_embed_kernel<<<T_TOK, blk, 0, stream>>>(input_ids, embed, xb);
  router_kernel<<<1, 32, 0, stream>>>(pool, w_router, gate, gidx);

  // 2) experts, sequential (reuse h buffer), combined into outf
  const dim3 g1(T_TOK / TM, HID / TN);   // (64, 32)
  const dim3 g2(T_TOK / TM, DIM / TN);   // (64, 8)
  for (int slot = 0; slot < 2; ++slot) {
    gemm_bf16_wmma_kernel<<<g1, blk, 0, stream>>>(
        xb, w1, (size_t)DIM * HID, gate, gidx, slot,
        hb, nullptr, HID, DIM, MODE_GELU_BF16, 0);
    gemm_bf16_wmma_kernel<<<g2, blk, 0, stream>>>(
        hb, w2, (size_t)HID * DIM, gate, gidx, slot,
        nullptr, outf, DIM, HID, MODE_SCALE_ACC_F32, slot);
  }

  // 3) head: convert combine to bf16, then (T x D) @ (D x V)
  f32_to_bf16_kernel<<<(T_TOK * DIM) / 1024, blk, 0, stream>>>(outf, outb);
  const dim3 g3(T_TOK / TM, VOC / TN);   // (64, 250)
  gemm_bf16_wmma_kernel<<<g3, blk, 0, stream>>>(
      outb, w_head, 0, gate, gidx, -1,
      nullptr, logits, VOC, DIM, MODE_STORE_F32, 0);
}
